// Attention_47751446397484
// MI455X (gfx1250) — matrix operand order repaired and checked by reading, not compiled, not run on hardware
//
#include <hip/hip_runtime.h>

// ---------------------------------------------------------------------------
// Types for CDNA5 WMMA (wave32, 16x16x32 bf16 -> f32)
// ---------------------------------------------------------------------------
typedef __bf16 bf16_t;
typedef bf16_t v16bf __attribute__((ext_vector_type(16)));
typedef float  v8f   __attribute__((ext_vector_type(8)));

struct FragBF {
    union {
        v16bf v;
        unsigned short u[16];
    };
};

__device__ __forceinline__ unsigned short f32_to_bf16(float f) {
    unsigned int u = __float_as_uint(f);
    unsigned int r = (u + 0x7FFFu + ((u >> 16) & 1u)) >> 16;
    return (unsigned short)r;
}

__device__ __forceinline__ v8f wmma_bf16(const FragBF& a, const FragBF& b, v8f c) {
    return __builtin_amdgcn_wmma_f32_16x16x32_bf16(
        /*neg_a=*/false, a.v, /*neg_b=*/false, b.v,
        /*c_mod=*/(short)0, c, /*reuse_a=*/false, /*reuse_b=*/false);
}

// ---------------------------------------------------------------------------
// CDNA5 async memory->LDS copy (CDNA5 ISA):
// GLOBAL_LOAD_ASYNC_TO_LDS_B128: VDST = LDS byte address (VGPR),
// VADDR = 64-bit global address (VGPR pair), GV mode. Tracked by ASYNCcnt.
// ---------------------------------------------------------------------------
#define USE_ASYNC_LDS 1

__device__ __forceinline__ void async_copy_b128(void* lds_dst, const void* gsrc) {
#if USE_ASYNC_LDS
    unsigned lds_off = (unsigned)(unsigned long long)lds_dst;   // addr[31:0] = LDS offset
    unsigned long long ga = (unsigned long long)gsrc;
    asm volatile("global_load_async_to_lds_b128 %0, %1, off"
                 :: "v"(lds_off), "v"(ga)
                 : "memory");
#else
    *(uint4*)lds_dst = *(const uint4*)gsrc;
#endif
}

__device__ __forceinline__ void wait_async_lds() {
#if USE_ASYNC_LDS
    asm volatile("s_wait_asynccnt 0" ::: "memory");
#endif
}

#define NEG_BIG (-3.0e38f)

// Problem constants (from reference)
#define BATCH   4
#define SEQ     2048
#define DIM     1024
#define HEADS   16
#define DHEAD   64
#define ROWS    (BATCH * SEQ)      // 8192
#define QKV_N   (3 * DIM)          // 3072

// ---------------------------------------------------------------------------
// Kernel 1: RMSNorm (F.normalize * sqrt(dim) * gamma) -> bf16
// ---------------------------------------------------------------------------
__global__ __launch_bounds__(256)
void rmsnorm_bf16(const float* __restrict__ x, const float* __restrict__ gamma,
                  unsigned short* __restrict__ xn) {
    const int row = blockIdx.x;
    const float* xr = x + (size_t)row * DIM;
    float ss = 0.0f;
    for (int c = threadIdx.x; c < DIM; c += 256) {
        float v = xr[c];
        ss += v * v;
    }
    for (int off = 16; off > 0; off >>= 1)
        ss += __shfl_xor(ss, off, 32);
    __shared__ float red[8];
    if ((threadIdx.x & 31) == 0) red[threadIdx.x >> 5] = ss;
    __syncthreads();
    float tot = 0.0f;
#pragma unroll
    for (int i = 0; i < 8; ++i) tot += red[i];
    const float inv = 32.0f / fmaxf(sqrtf(tot), 1e-12f);  // sqrt(1024) == 32
    for (int c = threadIdx.x; c < DIM; c += 256)
        xn[(size_t)row * DIM + c] = f32_to_bf16(xr[c] * inv * gamma[c]);
}

// ---------------------------------------------------------------------------
// Kernel 2: cast f32 -> bf16 with transpose: in[R][C] f32 -> out[C][R] bf16.
// LDS-tiled 32x32, 256 threads (32x8).
// ---------------------------------------------------------------------------
__global__ __launch_bounds__(256)
void cast_transpose_bf16(const float* __restrict__ in, unsigned short* __restrict__ out,
                         int R, int C) {
    __shared__ float tile[32][33];
    const int c0 = blockIdx.x * 32;
    const int r0 = blockIdx.y * 32;
    const int tx = threadIdx.x & 31;
    const int ty = threadIdx.x >> 5;   // 0..7
#pragma unroll
    for (int i = ty; i < 32; i += 8)
        tile[i][tx] = in[(size_t)(r0 + i) * C + c0 + tx];
    __syncthreads();
#pragma unroll
    for (int i = ty; i < 32; i += 8)
        out[(size_t)(c0 + i) * R + r0 + tx] = f32_to_bf16(tile[tx][i]);
}

// ---------------------------------------------------------------------------
// Kernel 3/5: bf16 GEMM, C[M,N] = A[M,K] * Bt[N,K]^T, fp32 accumulate.
// B is pre-transposed so BOTH operand fragments are contiguous ds_load_b128.
// Block: 256 threads = 8 waves; BM=128, BN=128, BK=32; wave -> 32x64 tile.
// Tiles staged with GLOBAL_LOAD_ASYNC_TO_LDS_B128 (ASYNCcnt).
// ---------------------------------------------------------------------------
template <bool OUT_BF16>
__global__ __launch_bounds__(256)
void gemm_bf16_wmma(const unsigned short* __restrict__ A,
                    const unsigned short* __restrict__ Bt,
                    void* __restrict__ C, int M, int N, int K) {
    constexpr int BM = 128, BN = 128, BK = 32;
    __shared__ __align__(16) unsigned short As[BM * BK];  // 8 KB  [row][k]
    __shared__ __align__(16) unsigned short Bs[BN * BK];  // 8 KB  [col][k]

    const int tid   = threadIdx.x;
    const int lane  = tid & 31;
    const int wave  = tid >> 5;
    const int wm    = wave >> 1;       // 0..3
    const int wn    = wave & 1;        // 0..1
    const int lhalf = lane >> 4;       // 0/1
    const int lrow  = lane & 15;
    const int block_m = blockIdx.x * BM;
    const int block_n = blockIdx.y * BN;

    v8f acc[2][4];
#pragma unroll
    for (int mi = 0; mi < 2; ++mi)
#pragma unroll
        for (int ni = 0; ni < 4; ++ni)
#pragma unroll
            for (int r = 0; r < 8; ++r) acc[mi][ni][r] = 0.0f;

    for (int k0 = 0; k0 < K; k0 += BK) {
        // Stage A tile (128x32) and Bt tile (128x32) async into LDS.
#pragma unroll
        for (int i = tid; i < (BM * BK) / 8; i += 256) {   // 512 chunks
            const int r = i >> 2;
            const int c = (i & 3) * 8;
            async_copy_b128(&As[r * BK + c],
                            &A[(size_t)(block_m + r) * K + k0 + c]);
            async_copy_b128(&Bs[r * BK + c],
                            &Bt[(size_t)(block_n + r) * K + k0 + c]);
        }
        // Prefetch next iteration's tiles toward L2
        if (k0 + BK < K) {
            __builtin_prefetch(&A[(size_t)(block_m + (tid >> 1)) * K + k0 + BK], 0, 1);
            __builtin_prefetch(&Bt[(size_t)(block_n + (tid >> 1)) * K + k0 + BK], 0, 1);
        }
        wait_async_lds();
        __syncthreads();

        // A fragments: ISA 16-bit A layout -> two contiguous 16B LDS reads
        FragBF a[2];
#pragma unroll
        for (int mi = 0; mi < 2; ++mi) {
            const int row = wm * 32 + mi * 16 + lrow;
            const unsigned short* p = &As[row * BK + lhalf * 8];
            *(uint4*)&a[mi].u[0] = *(const uint4*)p;         // K = lhalf*8 .. +7
            *(uint4*)&a[mi].u[8] = *(const uint4*)(p + 16);  // K = 16+lhalf*8 ..
        }
        // B fragments: K = lhalf*8 + s (s<8), 16 + lhalf*8 + (s-8) (s>=8), N = col
        FragBF b[4];
#pragma unroll
        for (int ni = 0; ni < 4; ++ni) {
            const int col = wn * 64 + ni * 16 + lrow;
            const unsigned short* p = &Bs[col * BK + lhalf * 8];
            *(uint4*)&b[ni].u[0] = *(const uint4*)p;
            *(uint4*)&b[ni].u[8] = *(const uint4*)(p + 16);
        }
#pragma unroll
        for (int mi = 0; mi < 2; ++mi)
#pragma unroll
            for (int ni = 0; ni < 4; ++ni)
                acc[mi][ni] = wmma_bf16(a[mi], b[ni], acc[mi][ni]);
        __syncthreads();
    }

    // Epilogue: D layout -> lane l, vgpr r holds (M = r + 8*(l>>4), N = l&15)
#pragma unroll
    for (int mi = 0; mi < 2; ++mi)
#pragma unroll
        for (int ni = 0; ni < 4; ++ni) {
            const int col = block_n + wn * 64 + ni * 16 + lrow;
#pragma unroll
            for (int r = 0; r < 8; ++r) {
                const int row = block_m + wm * 32 + mi * 16 + r + lhalf * 8;
                const float val = acc[mi][ni][r];
                if constexpr (OUT_BF16)
                    ((unsigned short*)C)[(size_t)row * N + col] = f32_to_bf16(val);
                else
                    ((float*)C)[(size_t)row * N + col] = val;
            }
        }
}

// ---------------------------------------------------------------------------
// Kernel 4: flash attention over the fused QKV buffer [rows, 3072] (bf16).
// Grid: (SEQ/64, HEADS, BATCH). Block: 128 threads = 4 waves x 16 query rows.
// K/V tiles staged async to LDS; online softmax fp32; q-scale folded in.
// ---------------------------------------------------------------------------
__global__ __launch_bounds__(128)
void flash_attn(const unsigned short* __restrict__ qkv,
                const unsigned char* __restrict__ mask,
                unsigned short* __restrict__ att) {
    const int qb   = blockIdx.x * 64;
    const int h    = blockIdx.y;
    const int bidx = blockIdx.z;
    const int tid  = threadIdx.x;
    const int lane = tid & 31;
    const int wave = tid >> 5;
    const int lhalf = lane >> 4;
    const int lrow  = lane & 15;

    __shared__ __align__(16) unsigned short Ks[64 * 64];       // 8 KB
    __shared__ __align__(16) unsigned short Vs[64 * 64];       // 8 KB
    __shared__ __align__(16) unsigned short Ps[4][16 * 64];    // 8 KB

    const size_t rs = QKV_N;
    const unsigned short* qp = qkv + (size_t)(bidx * SEQ) * rs + h * DHEAD;
    const unsigned short* kp = qp + DIM;
    const unsigned short* vp = qp + 2 * DIM;

    // Q fragments for this wave's 16 rows (A layout), dims 0..31 and 32..63
    FragBF qf[2];
    {
        const unsigned short* p = qp + (size_t)(qb + wave * 16 + lrow) * rs;
        *(uint4*)&qf[0].u[0] = *(const uint4*)(p + lhalf * 8);
        *(uint4*)&qf[0].u[8] = *(const uint4*)(p + 16 + lhalf * 8);
        *(uint4*)&qf[1].u[0] = *(const uint4*)(p + 32 + lhalf * 8);
        *(uint4*)&qf[1].u[8] = *(const uint4*)(p + 48 + lhalf * 8);
    }

    float m_i[8], l_i[8];
    v8f acc[4];
#pragma unroll
    for (int r = 0; r < 8; ++r) { m_i[r] = NEG_BIG; l_i[r] = 0.0f; }
#pragma unroll
    for (int ni = 0; ni < 4; ++ni)
#pragma unroll
        for (int r = 0; r < 8; ++r) acc[ni][r] = 0.0f;

    for (int kb = 0; kb < SEQ; kb += 64) {
        // Stage K and V tiles (64 x 64 bf16 each) asynchronously
#pragma unroll
        for (int i = tid; i < 512; i += 128) {
            const int r = i >> 3;
            const int c = (i & 7) * 8;
            async_copy_b128(&Ks[r * 64 + c], kp + (size_t)(kb + r) * rs + c);
            async_copy_b128(&Vs[r * 64 + c], vp + (size_t)(kb + r) * rs + c);
        }
        wait_async_lds();
        __syncthreads();

        // S = Q * K^T : 4 key-tiles x 2 k-steps over dhead
        v8f s[4];
#pragma unroll
        for (int ni = 0; ni < 4; ++ni) {
#pragma unroll
            for (int r = 0; r < 8; ++r) s[ni][r] = 0.0f;
            const int key = ni * 16 + lrow;
#pragma unroll
            for (int ks = 0; ks < 2; ++ks) {
                FragBF kf;  // B frag: K=dim component (contiguous), N=key
                const unsigned short* p = &Ks[key * 64 + ks * 32 + lhalf * 8];
                *(uint4*)&kf.u[0] = *(const uint4*)p;
                *(uint4*)&kf.u[8] = *(const uint4*)(p + 16);
                s[ni] = wmma_bf16(qf[ks], kf, s[ni]);
            }
        }

        // Online softmax: scale, mask, row max/sum across 16-lane half groups
        const unsigned char* mrow = mask + (size_t)bidx * SEQ + kb;
        float rowmax[8];
#pragma unroll
        for (int r = 0; r < 8; ++r) rowmax[r] = NEG_BIG;
#pragma unroll
        for (int ni = 0; ni < 4; ++ni) {
            const int key = ni * 16 + lrow;
            const bool ok = mrow[key] != 0;
#pragma unroll
            for (int r = 0; r < 8; ++r) {
                float sv = s[ni][r] * 0.125f;
                sv = ok ? sv : NEG_BIG;
                s[ni][r] = sv;
                rowmax[r] = fmaxf(rowmax[r], sv);
            }
        }
#pragma unroll
        for (int off = 1; off < 16; off <<= 1)
#pragma unroll
            for (int r = 0; r < 8; ++r)
                rowmax[r] = fmaxf(rowmax[r], __shfl_xor(rowmax[r], off, 32));

        float alpha[8], rowsum[8];
#pragma unroll
        for (int r = 0; r < 8; ++r) {
            const float mn = fmaxf(m_i[r], rowmax[r]);
            alpha[r] = __expf(m_i[r] - mn);
            m_i[r] = mn;
            rowsum[r] = 0.0f;
        }
#pragma unroll
        for (int ni = 0; ni < 4; ++ni)
#pragma unroll
            for (int r = 0; r < 8; ++r) {
                const float p = __expf(s[ni][r] - m_i[r]);
                s[ni][r] = p;
                rowsum[r] += p;
            }
#pragma unroll
        for (int off = 1; off < 16; off <<= 1)
#pragma unroll
            for (int r = 0; r < 8; ++r)
                rowsum[r] += __shfl_xor(rowsum[r], off, 32);
#pragma unroll
        for (int r = 0; r < 8; ++r) l_i[r] = l_i[r] * alpha[r] + rowsum[r];
#pragma unroll
        for (int ni = 0; ni < 4; ++ni)
#pragma unroll
            for (int r = 0; r < 8; ++r) acc[ni][r] *= alpha[r];

        // P: D-layout -> per-wave LDS -> A-layout fragments
#pragma unroll
        for (int ni = 0; ni < 4; ++ni)
#pragma unroll
            for (int r = 0; r < 8; ++r)
                Ps[wave][(r + lhalf * 8) * 64 + ni * 16 + lrow] =
                    f32_to_bf16(s[ni][r]);

        FragBF pa[2];
#pragma unroll
        for (int ks = 0; ks < 2; ++ks) {
            const unsigned short* p = &Ps[wave][lrow * 64 + ks * 32 + lhalf * 8];
            *(uint4*)&pa[ks].u[0] = *(const uint4*)p;
            *(uint4*)&pa[ks].u[8] = *(const uint4*)(p + 16);
        }
        // acc += P * V : B frag (K=key row, N=dim col) gathered from Vs
#pragma unroll
        for (int ni = 0; ni < 4; ++ni) {
            const int col = ni * 16 + lrow;
#pragma unroll
            for (int ks = 0; ks < 2; ++ks) {
                FragBF vf;
#pragma unroll
                for (int hh = 0; hh < 16; ++hh)
                    vf.u[hh] = Vs[(ks * 32 + lhalf * 8 + (hh & 7) + (hh >> 3) * 16) * 64 + col];
                acc[ni] = wmma_bf16(pa[ks], vf, acc[ni]);
            }
        }
        __syncthreads();
    }

    // Finalize: out = acc / l, store bf16 to [rows, h*64 + dim]
#pragma unroll
    for (int ni = 0; ni < 4; ++ni) {
        const int col = h * DHEAD + ni * 16 + lrow;
#pragma unroll
        for (int r = 0; r < 8; ++r) {
            const int rowg = bidx * SEQ + qb + wave * 16 + r + lhalf * 8;
            const float inv = 1.0f / l_i[r];
            att[(size_t)rowg * DIM + col] = f32_to_bf16(acc[ni][r] * inv);
        }
    }
}

// ---------------------------------------------------------------------------
// Host launcher
// d_in order: x[4,2048,1024] f32, mask[4,2048] bool, gamma[1024] f32,
//             w_qkv[1024,3072] f32, w_out[1024,1024] f32
// d_out: [4,2048,1024] f32
// ---------------------------------------------------------------------------
extern "C" void kernel_launch(void* const* d_in, const int* in_sizes, int n_in,
                              void* d_out, int out_size, void* d_ws, size_t ws_size,
                              hipStream_t stream) {
    (void)in_sizes; (void)n_in; (void)out_size; (void)ws_size;
    const float* x     = (const float*)d_in[0];
    const unsigned char* mask = (const unsigned char*)d_in[1];
    const float* gamma = (const float*)d_in[2];
    const float* wqkv  = (const float*)d_in[3];
    const float* wout  = (const float*)d_in[4];
    float* out = (float*)d_out;

    // Workspace layout (bf16 = ushort)
    unsigned short* xn_bf   = (unsigned short*)d_ws;                 // ROWS*DIM
    unsigned short* wqkv_t  = xn_bf + (size_t)ROWS * DIM;            // QKV_N*DIM
    unsigned short* wout_t  = wqkv_t + (size_t)DIM * QKV_N;          // DIM*DIM
    unsigned short* qkv_bf  = wout_t + (size_t)DIM * DIM;            // ROWS*QKV_N
    unsigned short* att_bf  = qkv_bf + (size_t)ROWS * QKV_N;         // ROWS*DIM

    // 1) RMSNorm + bf16 cast
    rmsnorm_bf16<<<ROWS, 256, 0, stream>>>(x, gamma, xn_bf);

    // 2) weight cast + transpose: w[K][N] -> wt[N][K] bf16
    {
        dim3 g1(QKV_N / 32, DIM / 32);
        cast_transpose_bf16<<<g1, 256, 0, stream>>>(wqkv, wqkv_t, DIM, QKV_N);
        dim3 g2(DIM / 32, DIM / 32);
        cast_transpose_bf16<<<g2, 256, 0, stream>>>(wout, wout_t, DIM, DIM);
    }

    // 3) QKV GEMM: [8192,1024] x [1024,3072] -> bf16
    {
        dim3 grid(ROWS / 128, QKV_N / 128);
        gemm_bf16_wmma<true><<<grid, 256, 0, stream>>>(
            xn_bf, wqkv_t, (void*)qkv_bf, ROWS, QKV_N, DIM);
    }

    // 4) Flash attention
    {
        dim3 grid(SEQ / 64, HEADS, BATCH);
        flash_attn<<<grid, 128, 0, stream>>>(qkv_bf, mask, att_bf);
    }

    // 5) Output projection: [8192,1024] x [1024,1024] -> f32
    {
        dim3 grid(ROWS / 128, DIM / 128);
        gemm_bf16_wmma<false><<<grid, 256, 0, stream>>>(
            att_bf, wout_t, (void*)out, ROWS, DIM, DIM);
    }
}
